// CausalSelfAttention_2680059593231
// MI455X (gfx1250) — matrix operand order repaired and checked by reading, not compiled, not run on hardware
//
#include <hip/hip_runtime.h>
#include <stdint.h>

// ---------------------------------------------------------------------------
// CausalSelfAttention for MI455X (gfx1250), wave32 + WMMA bf16 + TDM staging.
// Problem: B=2, L=2048, D=2048, H=16, DH=128.
// ---------------------------------------------------------------------------

typedef __attribute__((ext_vector_type(16))) __bf16   bf16x16;
typedef __attribute__((ext_vector_type(8)))  float    f32x8;
typedef __attribute__((ext_vector_type(8)))  uint16_t u16x8;
typedef __attribute__((ext_vector_type(16))) uint16_t u16x16;
typedef __attribute__((ext_vector_type(4)))  uint32_t u32x4;
typedef __attribute__((ext_vector_type(4)))  int32_t  i32x4;
typedef __attribute__((ext_vector_type(8)))  int32_t  i32x8;

static constexpr int CB  = 2;
static constexpr int CL  = 2048;
static constexpr int CD  = 2048;
static constexpr int CH  = 16;
static constexpr int CDH = 128;
static constexpr int CM  = CB * CL;     // 4096 rows
static constexpr int CN_QKV = 3 * CD;   // 6144

// Tensor Data Mover availability (device pass only; host pass falls back).
#if defined(__has_builtin)
#if __has_builtin(__builtin_amdgcn_tensor_load_to_lds) && \
    __has_builtin(__builtin_amdgcn_s_wait_tensorcnt)
#define HAVE_TDM 1
#if __has_include(<hip/amd_detail/amd_gfx1250_TDM.h>)
#define TDM_ARGS6 1
#endif
#endif
#endif

// ---------------- scalar helpers ----------------

__device__ __forceinline__ uint16_t f2bf(float f) {
  uint32_t u = __float_as_uint(f);
  u += 0x7FFFu + ((u >> 16) & 1u);       // round-to-nearest-even
  return (uint16_t)(u >> 16);
}
__device__ __forceinline__ float bf2f(uint16_t h) {
  return __uint_as_float(((uint32_t)h) << 16);
}

// ---------------- WMMA fragment helpers ----------------

// A-fragment (16x32 bf16): lane l<16 owns row m0+l with K 0..15;
// lane l>=16 owns row with K 16..31. rowptr = &A[(m0+(lane&15))*ld + k0].
__device__ __forceinline__ bf16x16 load_a_frag(const uint16_t* rowptr, int half) {
  u16x8 lo = *(const u16x8*)(rowptr + half * 16);
  u16x8 hi = *(const u16x8*)(rowptr + half * 16 + 8);
  u16x16 w = __builtin_shufflevector(lo, hi, 0,1,2,3,4,5,6,7,8,9,10,11,12,13,14,15);
  return __builtin_bit_cast(bf16x16, w);
}

// B-fragment (32x16 bf16): lane l<16 owns column n0+l with K 0..15, lane l>=16
// owns same column set with K 16..31. p = &W[(n0+(lane&15))*ld + k0 + half*16].
__device__ __forceinline__ bf16x16 load_b_frag(const uint16_t* p) {
  return __builtin_bit_cast(bf16x16, *(const u16x16*)p);
}

__device__ __forceinline__ f32x8 wmma_bf16(bf16x16 a, bf16x16 b, f32x8 c) {
  return __builtin_amdgcn_wmma_f32_16x16x32_bf16(false, a, false, b, (short)0, c,
                                                 false, false);
}

// ---------------- Tensor Data Mover: 2D tile -> LDS ----------------
// Packs D# group0/group1 per CDNA5 ISA §8 (count=1, type=2 "image",
// data_size=1 -> 2 bytes, 2D tile, dim0 stride in elements).

#if defined(HAVE_TDM)
__device__ __forceinline__ void tdm_load_2d(uint32_t lds_addr, const void* gaddr,
                                            uint32_t tile_w, uint32_t tile_h,
                                            uint32_t row_stride) {
  const uint64_t ga = (uint64_t)(uintptr_t)gaddr;
  u32x4 g0;
  g0[0] = 1u;                                       // count=1, user-mode
  g0[1] = lds_addr;                                 // LDS byte address
  g0[2] = (uint32_t)ga;                             // global_addr[31:0]
  g0[3] = (uint32_t)((ga >> 32) & 0x01FFFFFFu)      // global_addr[56:32]
          | (2u << 30);                             // type = 2
  i32x8 g1;
  g1[0] = (int32_t)(1u << 16);                      // data_size=1 (2B), mask=0
  g1[1] = (int32_t)(tile_w << 16);                  // tensor_dim0[15:0]
  g1[2] = (int32_t)((tile_w >> 16) | (tile_h << 16)); // dim0[31:16] | dim1[15:0]
  g1[3] = (int32_t)((tile_h >> 16) | (tile_w << 16)); // dim1[31:16] | tile_dim0
  g1[4] = (int32_t)(tile_h & 0xFFFFu);              // tile_dim1 | tile_dim2=0
  g1[5] = (int32_t)row_stride;                      // tensor_dim0_stride[31:0]
  g1[6] = 0;                                        // stride[47:32] | dim1_stride lo
  g1[7] = 0;
  i32x4 z4 = {0, 0, 0, 0};
#if defined(TDM_ARGS6)
  i32x8 z8 = {0, 0, 0, 0, 0, 0, 0, 0};
  __builtin_amdgcn_tensor_load_to_lds(g0, g1, z4, z4, z8, 0);
#else
  __builtin_amdgcn_tensor_load_to_lds(g0, g1, z4, z4, 0);
#endif
}
#endif

// ---------------- shared GEMM main loop ----------------
// Block = 8 waves; block tile M=128 (wave-owns-16-rows), N=64.
// acc[t] accumulates the wave's 16x16 tile at column t*16.
// TDM path: double-buffered 64x64 bf16 W-chunks staged in LDS by wave 0;
// uniform K loop across waves so block barriers are safe.
// A-fragments are software-pipelined one chunk ahead; all 8 B-fragments of a
// chunk are loaded before the 8 WMMAs so the XDL pipe issues back-to-back.

#define GEMM_LDS_CHUNK (64 * 64)   // elements per buffer (8 KB)

__device__ __forceinline__ void gemm_mainloop(const uint16_t* __restrict__ arow,
                                              const uint16_t* __restrict__ wtile,
                                              uint16_t* ldsW, int K,
                                              int wave, int ln, int half,
                                              f32x8 acc[4]) {
#if defined(HAVE_TDM)
  if (wave == 0)
    tdm_load_2d((uint32_t)(uintptr_t)ldsW, wtile, 64u, 64u, (uint32_t)K);
  // A-fragment pipeline: chunk 0 preloaded before the loop.
  bf16x16 a0 = load_a_frag(arow, half);
  bf16x16 a1 = load_a_frag(arow + 32, half);
  for (int k0 = 0; k0 < K; k0 += 64) {
    if (wave == 0) __builtin_amdgcn_s_wait_tensorcnt(0);
    __syncthreads();                                   // publish current buffer
    const int cb = (k0 >> 6) & 1;
    const uint16_t* cur = ldsW + cb * GEMM_LDS_CHUNK;
    if (wave == 0 && (k0 + 64) < K)                    // stream next chunk now
      tdm_load_2d((uint32_t)(uintptr_t)(ldsW + (cb ^ 1) * GEMM_LDS_CHUNK),
                  wtile + k0 + 64, 64u, 64u, (uint32_t)K);
    // A-fragments for NEXT chunk issued before this chunk's WMMAs.
    bf16x16 na0 = a0, na1 = a1;
    if (k0 + 64 < K) {
      __builtin_prefetch(arow + k0 + 512, 0, 3);
      na0 = load_a_frag(arow + k0 + 64, half);
      na1 = load_a_frag(arow + k0 + 96, half);
    }
    // Batch all 8 B-fragments (ds_load clauses), then 8 back-to-back WMMAs.
    bf16x16 bfr[8];
#pragma unroll
    for (int t = 0; t < 4; ++t) {
      const uint16_t* lp = cur + (t * 16 + ln) * 64;
      bfr[2 * t]     = load_b_frag(lp + half * 16);
      bfr[2 * t + 1] = load_b_frag(lp + 32 + half * 16);
    }
#pragma unroll
    for (int t = 0; t < 4; ++t) {
      acc[t] = wmma_bf16(a0, bfr[2 * t], acc[t]);
      acc[t] = wmma_bf16(a1, bfr[2 * t + 1], acc[t]);
    }
    a0 = na0; a1 = na1;
    __syncthreads();                                   // retire current buffer
  }
#else
  (void)ldsW; (void)wave;
  for (int k0 = 0; k0 < K; k0 += 32) {
    __builtin_prefetch(arow + k0 + 256, 0, 3);
    bf16x16 a = load_a_frag(arow + k0, half);
    bf16x16 bfr[4];
#pragma unroll
    for (int t = 0; t < 4; ++t)
      bfr[t] = load_b_frag(wtile + (size_t)(t * 16 + ln) * K + k0 + half * 16);
#pragma unroll
    for (int t = 0; t < 4; ++t) acc[t] = wmma_bf16(a, bfr[t], acc[t]);
  }
#endif
}

// ---------------- kernels ----------------

__global__ void cvt_bf16_kernel(const float* __restrict__ src,
                                uint16_t* __restrict__ dst, int n) {
  int i = blockIdx.x * blockDim.x + threadIdx.x;
  if (i < n) dst[i] = f2bf(src[i]);
}

// QKV projection: qkv[m, e] = sum_d x[m, d] * w_qkv[e, d]
// Epilogue scatters into q[b,h,l,dh], k[b,h,l,dh], vT[b,h,dh,l] (bf16).
__global__ __launch_bounds__(256)
void qkv_gemm_kernel(const uint16_t* __restrict__ xh,
                     const uint16_t* __restrict__ wh,
                     uint16_t* __restrict__ qd,
                     uint16_t* __restrict__ kd,
                     uint16_t* __restrict__ vT) {
  __shared__ __align__(128) uint16_t ldsW[2 * GEMM_LDS_CHUNK];
  const int lane = threadIdx.x & 31;
  const int wave = threadIdx.x >> 5;
  const int half = lane >> 4;
  const int ln   = lane & 15;
  const int mbase = blockIdx.y * 128 + wave * 16;
  const int nbase = blockIdx.x * 64;
  const int K = CD;

  const f32x8 zero = {0.f,0.f,0.f,0.f,0.f,0.f,0.f,0.f};
  f32x8 acc[4] = {zero, zero, zero, zero};

  const uint16_t* arow = xh + (size_t)(mbase + ln) * K;
  gemm_mainloop(arow, wh + (size_t)nbase * K, ldsW, K, wave, ln, half, acc);

#pragma unroll
  for (int t = 0; t < 4; ++t) {
    const int n     = nbase + t * 16 + ln;
    const int which = n >> 11;            // 0=q 1=k 2=v
    const int rem   = n & 2047;
    const int hh    = rem >> 7;
    const int d     = rem & 127;
#pragma unroll
    for (int r = 0; r < 8; ++r) {
      const int m  = mbase + r + 8 * half;
      const int bb = m >> 11;
      const int l  = m & 2047;
      const uint16_t val = f2bf(acc[t][r]);
      if (which == 0)
        qd[(((size_t)bb * CH + hh) * CL + l) * CDH + d] = val;
      else if (which == 1)
        kd[(((size_t)bb * CH + hh) * CL + l) * CDH + d] = val;
      else
        vT[(((size_t)bb * CH + hh) * CDH + d) * CL + l] = val;
    }
  }
}

// Half-split RoPE applied in place to a [B,H,L,DH] bf16 tensor.
__global__ void rope_kernel(uint16_t* __restrict__ t, int total) {
  int tid = blockIdx.x * blockDim.x + threadIdx.x;
  if (tid >= total) return;
  const int i  = tid & 63;                // pair index 0..63
  const int l  = (tid >> 6) & (CL - 1);
  const int bh = tid >> 17;               // b*H + h
  const size_t base = ((size_t)bh * CL + l) * CDH;
  const float inv_freq = __powf(10000.0f, -(float)(2 * i) / (float)CDH);
  float s, c;
  __sincosf((float)l * inv_freq, &s, &c);
  const float x1 = bf2f(t[base + i]);
  const float x2 = bf2f(t[base + i + 64]);
  t[base + i]      = f2bf(x1 * c - x2 * s);
  t[base + i + 64] = f2bf(x2 * c + x1 * s);
}

// Flash attention, causal. Grid (L/128, H, B), 256 thr = 8 waves.
// Each wave owns 16 query rows; loops 32-key tiles; online softmax in f32.
__global__ __launch_bounds__(256)
void flash_attn_kernel(const uint16_t* __restrict__ qd,
                       const uint16_t* __restrict__ kd,
                       const uint16_t* __restrict__ vT,
                       uint16_t* __restrict__ attn) {
  __shared__ __align__(32) uint16_t ldsP[8][16][32];   // per-wave P staging

  const int lane = threadIdx.x & 31;
  const int wave = threadIdx.x >> 5;
  const int half = lane >> 4;
  const int ln   = lane & 15;
  const int h = blockIdx.y, b = blockIdx.z;
  const int qbase = blockIdx.x * 128 + wave * 16;

  const uint16_t* qp = qd + (((size_t)b * CH + h) * CL) * CDH;
  const uint16_t* kp = kd + (((size_t)b * CH + h) * CL) * CDH;
  const uint16_t* vp = vT + (((size_t)b * CH + h) * CDH) * CL;

  // Preload all 4 Q A-fragments (16 rows x 128 dh)
  bf16x16 aq[4];
  {
    const uint16_t* qrow = qp + (size_t)(qbase + ln) * CDH;
#pragma unroll
    for (int kk = 0; kk < 4; ++kk) aq[kk] = load_a_frag(qrow + kk * 32, half);
  }

  const f32x8 zero = {0.f,0.f,0.f,0.f,0.f,0.f,0.f,0.f};
  f32x8 o[8];
#pragma unroll
  for (int t = 0; t < 8; ++t) o[t] = zero;
  float mrow[8], lrow[8];
#pragma unroll
  for (int r = 0; r < 8; ++r) { mrow[r] = -3.0e38f; lrow[r] = 0.0f; }

  const float scale = 0.08838834764831845f;  // 1/sqrt(128)
  const int qmax = qbase + 15;

  for (int j = 0; j <= qmax; j += 32) {
    // ---- S = Q * K^T : two 16x16 f32 tiles (cols j..j+15, j+16..j+31).
    // Load 4 B-fragments per pair of K-steps, then 4 back-to-back WMMAs.
    f32x8 s0 = zero, s1 = zero;
#pragma unroll
    for (int kk = 0; kk < 4; kk += 2) {
      bf16x16 b00 = load_b_frag(kp + (size_t)(j + ln) * CDH + kk * 32 + half * 16);
      bf16x16 b10 = load_b_frag(kp + (size_t)(j + 16 + ln) * CDH + kk * 32 + half * 16);
      bf16x16 b01 = load_b_frag(kp + (size_t)(j + ln) * CDH + (kk + 1) * 32 + half * 16);
      bf16x16 b11 = load_b_frag(kp + (size_t)(j + 16 + ln) * CDH + (kk + 1) * 32 + half * 16);
      s0 = wmma_bf16(aq[kk], b00, s0);
      s1 = wmma_bf16(aq[kk], b10, s1);
      s0 = wmma_bf16(aq[kk + 1], b01, s0);
      s1 = wmma_bf16(aq[kk + 1], b11, s1);
    }

    // ---- causal mask + scale (C layout: row r+8*half, col (j+)ln / +16)
    const int c0 = j + ln, c1 = c0 + 16;
#pragma unroll
    for (int r = 0; r < 8; ++r) {
      const int row = qbase + r + 8 * half;
      s0[r] = (c0 <= row) ? s0[r] * scale : -3.0e38f;
      s1[r] = (c1 <= row) ? s1[r] * scale : -3.0e38f;
    }

    // ---- online softmax stats (row spread over 16 lanes of a half-wave)
#pragma unroll
    for (int r = 0; r < 8; ++r) {
      float tmax = fmaxf(s0[r], s1[r]);
#pragma unroll
      for (int off = 8; off > 0; off >>= 1)
        tmax = fmaxf(tmax, __shfl_xor(tmax, off, 32));
      const float mnew  = fmaxf(mrow[r], tmax);
      const float alpha = __expf(mrow[r] - mnew);
      const float p0 = __expf(s0[r] - mnew);
      const float p1 = __expf(s1[r] - mnew);
      s0[r] = p0; s1[r] = p1;
      float ps = p0 + p1;
#pragma unroll
      for (int off = 8; off > 0; off >>= 1) ps += __shfl_xor(ps, off, 32);
      lrow[r] = lrow[r] * alpha + ps;
      mrow[r] = mnew;
#pragma unroll
      for (int t = 0; t < 8; ++t) o[t][r] *= alpha;
    }

    // ---- C-layout -> A-layout for P via per-wave LDS (same-wave RAW;
    // no block barrier — per-wave buffer, divergent trip counts)
#pragma unroll
    for (int r = 0; r < 8; ++r) {
      ldsP[wave][r + 8 * half][ln]      = f2bf(s0[r]);
      ldsP[wave][r + 8 * half][16 + ln] = f2bf(s1[r]);
    }
    bf16x16 ap = load_a_frag(&ldsP[wave][ln][0], half);

    // ---- O += P(16x32) * V(32x128); V^T makes B columns contiguous.
    // Pairs of fragment loads then back-to-back WMMAs.
#pragma unroll
    for (int t = 0; t < 8; t += 2) {
      bf16x16 bv0 = load_b_frag(vp + (size_t)(t * 16 + ln) * CL + j + half * 16);
      bf16x16 bv1 = load_b_frag(vp + (size_t)((t + 1) * 16 + ln) * CL + j + half * 16);
      o[t]     = wmma_bf16(ap, bv0, o[t]);
      o[t + 1] = wmma_bf16(ap, bv1, o[t + 1]);
    }
  }

  // ---- normalize and write attn_out[b, l, h*DH + d] (bf16)
  float inv[8];
#pragma unroll
  for (int r = 0; r < 8; ++r) inv[r] = 1.0f / lrow[r];
#pragma unroll
  for (int t = 0; t < 8; ++t) {
#pragma unroll
    for (int r = 0; r < 8; ++r) {
      const int l = qbase + r + 8 * half;
      attn[((size_t)(b * CL + l)) * CD + h * CDH + t * 16 + ln] =
          f2bf(o[t][r] * inv[r]);
    }
  }
}

// Output projection: out[m, e] = sum_d attn[m, d] * w_out[e, d], f32 result.
__global__ __launch_bounds__(256)
void out_gemm_kernel(const uint16_t* __restrict__ ah,
                     const uint16_t* __restrict__ wh,
                     float* __restrict__ out) {
  __shared__ __align__(128) uint16_t ldsW[2 * GEMM_LDS_CHUNK];
  const int lane = threadIdx.x & 31;
  const int wave = threadIdx.x >> 5;
  const int half = lane >> 4;
  const int ln   = lane & 15;
  const int mbase = blockIdx.y * 128 + wave * 16;
  const int nbase = blockIdx.x * 64;
  const int K = CD;

  const f32x8 zero = {0.f,0.f,0.f,0.f,0.f,0.f,0.f,0.f};
  f32x8 acc[4] = {zero, zero, zero, zero};

  const uint16_t* arow = ah + (size_t)(mbase + ln) * K;
  gemm_mainloop(arow, wh + (size_t)nbase * K, ldsW, K, wave, ln, half, acc);

#pragma unroll
  for (int t = 0; t < 4; ++t) {
    const int n = nbase + t * 16 + ln;
#pragma unroll
    for (int r = 0; r < 8; ++r) {
      const int m = mbase + r + 8 * half;
      out[(size_t)m * CD + n] = acc[t][r];
    }
  }
}

// ---------------- host launcher ----------------

extern "C" void kernel_launch(void* const* d_in, const int* in_sizes, int n_in,
                              void* d_out, int out_size, void* d_ws, size_t ws_size,
                              hipStream_t stream) {
  (void)in_sizes; (void)n_in; (void)out_size; (void)ws_size;
  const float* x     = (const float*)d_in[0];
  const float* w_qkv = (const float*)d_in[1];
  const float* w_out = (const float*)d_in[2];
  float* out = (float*)d_out;

  uint16_t* ws = (uint16_t*)d_ws;
  size_t off = 0;
  const size_t SZ_X  = (size_t)CM * CD;             // 8.39M
  const size_t SZ_WQ = (size_t)CN_QKV * CD;         // 12.6M
  const size_t SZ_WO = (size_t)CD * CD;             // 4.19M
  const size_t SZ_T  = (size_t)CB * CH * CL * CDH;  // 8.39M

  uint16_t* xh  = ws + off; off += SZ_X;
  uint16_t* wqh = ws + off; off += SZ_WQ;
  uint16_t* woh = ws + off; off += SZ_WO;
  uint16_t* qd  = ws + off; off += SZ_T;
  uint16_t* kd  = ws + off; off += SZ_T;
  uint16_t* vT  = ws + off; off += SZ_T;
  uint16_t* att = ws + off; off += SZ_T;

  cvt_bf16_kernel<<<(int)((SZ_X  + 255) / 256), 256, 0, stream>>>(x, xh, (int)SZ_X);
  cvt_bf16_kernel<<<(int)((SZ_WQ + 255) / 256), 256, 0, stream>>>(w_qkv, wqh, (int)SZ_WQ);
  cvt_bf16_kernel<<<(int)((SZ_WO + 255) / 256), 256, 0, stream>>>(w_out, woh, (int)SZ_WO);

  qkv_gemm_kernel<<<dim3(CN_QKV / 64, CM / 128), 256, 0, stream>>>(xh, wqh, qd, kd, vT);

  const int rope_total = CB * CH * CL * (CDH / 2);  // 4.19M pair-threads
  rope_kernel<<<(rope_total + 255) / 256, 256, 0, stream>>>(qd, rope_total);
  rope_kernel<<<(rope_total + 255) / 256, 256, 0, stream>>>(kd, rope_total);

  flash_attn_kernel<<<dim3(CL / 128, CH, CB), 256, 0, stream>>>(qd, kd, vT, att);

  out_gemm_kernel<<<dim3(CD / 64, CM / 128), 256, 0, stream>>>(att, woh, out);
}
